// MLA_26293789786501
// MI455X (gfx1250) — matrix operand order repaired and checked by reading, not compiled, not run on hardware
//
#include <hip/hip_runtime.h>
#include <hip/hip_bf16.h>

typedef __bf16 bf16;
typedef __attribute__((ext_vector_type(16))) __bf16 bf16x16;
typedef __attribute__((ext_vector_type(8)))  float  f32x8;

#define B_    2
#define S_    2048
#define DIM_  2048
#define NH_   16
#define KVR_  512
#define NOPE_ 128
#define ROPE_ 64
#define VHD_  128
#define QKHD_ 192
#define KEFF_ 576
#define SCALE_ 0.07216878364870323f   // 192^-0.5

// ---------------------------------------------------------------------------
// CDNA5 async global->LDS copy (ASYNCcnt-tracked DMA path, ISA §10/§15.18.3).
// Each lane moves 16 bytes from its global address to its LDS address.
// The low 32 bits of a generic pointer to __shared__ are the wave-relative
// LDS byte address (flat aperture rule), which is exactly what VDST wants.
// ---------------------------------------------------------------------------
__device__ __forceinline__ void async_copy16(const bf16* g, bf16* l) {
    unsigned lds = (unsigned)(unsigned long long)l;
    unsigned long long ga = (unsigned long long)g;
    asm volatile("global_load_async_to_lds_b128 %0, %1, off"
                 :: "v"(lds), "v"(ga) : "memory");
}
__device__ __forceinline__ void async_wait0() {
    asm volatile("s_wait_asynccnt 0x0" ::: "memory");
}

// ---------------------------------------------------------------------------
// fp32 -> bf16 cast (grid-stride)
// ---------------------------------------------------------------------------
__global__ void cast_f32_bf16(const float* __restrict__ in, bf16* __restrict__ out, long n) {
    long i = (long)blockIdx.x * blockDim.x + threadIdx.x;
    long stride = (long)gridDim.x * blockDim.x;
    for (; i < n; i += stride) out[i] = (bf16)in[i];
}

// ---------------------------------------------------------------------------
// Generic tiled WMMA GEMM: C = A(MxK) * B,  B either (N,K) row-major (bIsNT=1,
// i.e. C = A*B^T) or (K,N) row-major (bIsNT=0, C = A*B).
// Batched over blockIdx.z = zb*inner + zh with independent strides.
// Tiles: BM=64, BN=64, BK=32; 128 threads = 4 waves, each wave a 32x32 subtile.
// ---------------------------------------------------------------------------
__global__ __launch_bounds__(128) void wmma_gemm(
    const bf16* __restrict__ A, const bf16* __restrict__ Bm, void* __restrict__ Cv,
    int M, int N, int K,
    long lda, long ldb, long ldc,
    long aSb, long aSh, long bSb, long bSh, long cSb, long cSh,
    int inner, int bIsNT, int outBF16)
{
    int z  = blockIdx.z;
    int zb = z / inner, zh = z % inner;
    const bf16* Ap = A  + (size_t)zb * aSb + (size_t)zh * aSh;
    const bf16* Bp = Bm + (size_t)zb * bSb + (size_t)zh * bSh;
    size_t cOff    = (size_t)zb * cSb + (size_t)zh * cSh;

    __shared__ bf16 sA[64][32];   // [m][k]
    __shared__ bf16 sB[64][32];   // [n][k]

    int tid  = threadIdx.x;
    int wave = tid >> 5, lane = tid & 31, half = lane >> 4, lp = lane & 15;
    int wm = wave >> 1, wn = wave & 1;
    int mBase = blockIdx.y * 64, nBase = blockIdx.x * 64;

    f32x8 acc[2][2];
    for (int i = 0; i < 2; i++)
        for (int j = 0; j < 2; j++)
            for (int e = 0; e < 8; e++) acc[i][j][e] = 0.f;

    for (int k0 = 0; k0 < K; k0 += 32) {
        __syncthreads();
        {   // stage A: each thread DMAs 2x16B straight into LDS
            int r = tid >> 1, hs = tid & 1;
            const bf16* g = Ap + (size_t)(mBase + r) * lda + k0 + hs * 16;
            async_copy16(g,     &sA[r][hs * 16]);
            async_copy16(g + 8, &sA[r][hs * 16 + 8]);
            if (k0 + 32 < K)
                __builtin_prefetch((const void*)(g + 32), 0, 3);
        }
        if (bIsNT) {
            int n = tid >> 1, hs = tid & 1;
            const bf16* g = Bp + (size_t)(nBase + n) * ldb + k0 + hs * 16;
            async_copy16(g,     &sB[n][hs * 16]);
            async_copy16(g + 8, &sB[n][hs * 16 + 8]);
        } else {
            int kR = tid >> 2, q4 = tid & 3;
            const bf16* src = Bp + (size_t)(k0 + kR) * ldb + nBase + q4 * 16;
            bf16 tmp[16];
            *(uint4*)&tmp[0] = ((const uint4*)src)[0];
            *(uint4*)&tmp[8] = ((const uint4*)src)[1];
            for (int j = 0; j < 16; j++) sB[q4 * 16 + j][kR] = tmp[j];
        }
        async_wait0();
        __syncthreads();

        bf16x16 af[2], bfr[2];
        for (int mi = 0; mi < 2; mi++) {
            int m = wm * 32 + mi * 16 + lp;
            for (int j = 0; j < 16; j++) {
                int kk = (j < 8) ? (half * 8 + j) : (16 + half * 8 + (j - 8));
                af[mi][j] = sA[m][kk];
            }
        }
        for (int ni = 0; ni < 2; ni++) {
            int n = wn * 32 + ni * 16 + lp;
            for (int j = 0; j < 16; j++) bfr[ni][j] = sB[n][(j < 8) ? (half * 8 + j) : (16 + half * 8 + (j - 8))];
        }
        for (int mi = 0; mi < 2; mi++)
            for (int ni = 0; ni < 2; ni++)
                acc[mi][ni] = __builtin_amdgcn_wmma_f32_16x16x32_bf16(
                    false, af[mi], false, bfr[ni], (short)0, acc[mi][ni], false, false);
    }

    for (int mi = 0; mi < 2; mi++)
        for (int ni = 0; ni < 2; ni++)
            for (int e = 0; e < 8; e++) {
                int r = mBase + wm * 32 + mi * 16 + e + half * 8;
                int c = nBase + wn * 32 + ni * 16 + lp;
                if (r < M && c < N) {
                    if (outBF16) ((bf16*)Cv)[cOff + (size_t)r * ldc + c] = (bf16)acc[mi][ni][e];
                    else         ((float*)Cv)[cOff + (size_t)r * ldc + c] = acc[mi][ni][e];
                }
            }
}

// ---------------------------------------------------------------------------
// kv epilogue: rms_norm on kv_c[0:512], RoPE on k_pe[512:576] -> k_eff bf16
// one block (128 thr) per token row
// ---------------------------------------------------------------------------
__global__ __launch_bounds__(128) void kv_finish(
    const float* __restrict__ kvF, const float* __restrict__ wNorm,
    const float* __restrict__ fCos, const float* __restrict__ fSin,
    bf16* __restrict__ kEff)
{
    int row = blockIdx.x;
    int s = row % S_;
    int tid = threadIdx.x;
    __shared__ float red[128];
    const float* src = kvF + (size_t)row * KEFF_;
    float ss = 0.f;
    for (int c = tid; c < KVR_; c += 128) { float v = src[c]; ss += v * v; }
    red[tid] = ss; __syncthreads();
    for (int off = 64; off > 0; off >>= 1) {
        if (tid < off) red[tid] += red[tid + off];
        __syncthreads();
    }
    float rn = rsqrtf(red[0] / (float)KVR_ + 1e-6f);
    for (int c = tid; c < KVR_; c += 128)
        kEff[(size_t)row * KEFF_ + c] = (bf16)(src[c] * rn * wNorm[c]);
    if (tid < 32) {
        int t = tid;
        float xr = src[KVR_ + 2 * t], xi = src[KVR_ + 2 * t + 1];
        float c = fCos[s * 32 + t], sn = fSin[s * 32 + t];
        kEff[(size_t)row * KEFF_ + KVR_ + 2 * t]     = (bf16)(xr * c - xi * sn);
        kEff[(size_t)row * KEFF_ + KVR_ + 2 * t + 1] = (bf16)(xr * sn + xi * c);
    }
}

// ---------------------------------------------------------------------------
// q epilogue: cast q_nope -> q_bf (GEMM A input), RoPE q_pe -> q_eff[...,512:576]
// grid (row=4096, h=16), 64 threads
// ---------------------------------------------------------------------------
__global__ __launch_bounds__(64) void q_finish(
    const float* __restrict__ qF, const float* __restrict__ fCos, const float* __restrict__ fSin,
    bf16* __restrict__ qBf, bf16* __restrict__ qEff)
{
    int row = blockIdx.x, h = blockIdx.y;
    int b = row / S_, s = row % S_;
    int tid = threadIdx.x;
    const float* src = qF + (size_t)row * (NH_ * QKHD_) + h * QKHD_;
    bf16* dstN = qBf + (size_t)row * (NH_ * QKHD_) + h * QKHD_;
    dstN[tid]      = (bf16)src[tid];
    dstN[tid + 64] = (bf16)src[tid + 64];
    if (tid < 32) {
        float xr = src[NOPE_ + 2 * tid], xi = src[NOPE_ + 2 * tid + 1];
        float c = fCos[s * 32 + tid], sn = fSin[s * 32 + tid];
        bf16* pe = qEff + ((size_t)(b * NH_ + h) * S_ + s) * KEFF_ + KVR_;
        pe[2 * tid]     = (bf16)(xr * c - xi * sn);
        pe[2 * tid + 1] = (bf16)(xr * sn + xi * c);
    }
}

// ---------------------------------------------------------------------------
// Flash attention over the latent cache.
// q_eff (B,NH,S,576), k_eff (B,S,576) where cols 0:512 double as V.
// grid (S/16, NH, B); 128 threads = 4 waves; wave w owns V cols [128w,128w+128).
// ---------------------------------------------------------------------------
__global__ __launch_bounds__(128) void mla_attn(
    const bf16* __restrict__ qEff, const bf16* __restrict__ kEff,
    bf16* __restrict__ oLat)
{
    int qt = blockIdx.x, h = blockIdx.y, b = blockIdx.z;
    int tid = threadIdx.x, wave = tid >> 5, lane = tid & 31, half = lane >> 4, lp = lane & 15;

    __shared__ bf16  sQ[16 * KEFF_];
    __shared__ bf16  sK[64][32];
    __shared__ float sS[16][64];
    __shared__ bf16  sP[16][64];
    __shared__ float sm[16], sl[16], sa[16];

    {   // Q tile is one contiguous 16x576 chunk; DMA it into LDS
        const bf16* src = qEff + ((size_t)(b * NH_ + h) * S_ + qt * 16) * KEFF_;
        for (int i = tid; i < 16 * KEFF_ / 8; i += 128)
            async_copy16(src + (size_t)i * 8, &sQ[i * 8]);
    }
    if (tid < 16) { sm[tid] = -__builtin_inff(); sl[tid] = 0.f; }

    f32x8 oacc[8];
    for (int nt = 0; nt < 8; nt++)
        for (int e = 0; e < 8; e++) oacc[nt][e] = 0.f;

    int iMax = qt * 16 + 15;
    int ktMax = iMax / 64;
    for (int kt = 0; kt <= ktMax; kt++) {
        int ktb = kt * 64;
        f32x8 sacc; for (int e = 0; e < 8; e++) sacc[e] = 0.f;

        // scores: each wave computes a 16x16 column strip of the 16x64 S tile
        for (int kk = 0; kk < KEFF_ / 32; kk++) {
            __syncthreads();
            {
                int kr = tid >> 1, hs = tid & 1;
                const bf16* g = kEff + ((size_t)b * S_ + ktb + kr) * KEFF_ + kk * 32 + hs * 16;
                async_copy16(g,     &sK[kr][hs * 16]);
                async_copy16(g + 8, &sK[kr][hs * 16 + 8]);
                if (kk + 1 < KEFF_ / 32)
                    __builtin_prefetch((const void*)(g + 32), 0, 3);
            }
            async_wait0();
            __syncthreads();
            bf16x16 af, bfr;
            for (int j = 0; j < 16; j++) {
                int kkl = (j < 8) ? (half * 8 + j) : (16 + half * 8 + (j - 8));
                af[j] = sQ[lp * KEFF_ + kk * 32 + kkl];
            }
            for (int j = 0; j < 16; j++) bfr[j] = sK[wave * 16 + lp][(j < 8) ? (half * 8 + j) : (16 + half * 8 + (j - 8))];
            sacc = __builtin_amdgcn_wmma_f32_16x16x32_bf16(
                false, af, false, bfr, (short)0, sacc, false, false);
        }

        // scale + causal mask -> sS
        for (int e = 0; e < 8; e++) {
            int r = e + half * 8;
            int i = qt * 16 + r;
            int j = ktb + wave * 16 + lp;
            float v = sacc[e] * SCALE_;
            if (j > i) v = -__builtin_inff();
            sS[r][wave * 16 + lp] = v;
        }
        __syncthreads();

        // online softmax row update (16 rows, one thread each)
        if (tid < 16) {
            int r = tid;
            float tm = -__builtin_inff();
            for (int j = 0; j < 64; j++) tm = fmaxf(tm, sS[r][j]);
            float mOld = sm[r];
            float mNew = fmaxf(mOld, tm);
            float alpha = __expf(mOld - mNew);
            float sum = 0.f;
            for (int j = 0; j < 64; j++) {
                float p = __expf(sS[r][j] - mNew);
                sP[r][j] = (bf16)p;
                sum += p;
            }
            sm[r] = mNew; sl[r] = sl[r] * alpha + sum; sa[r] = alpha;
        }
        __syncthreads();

        // rescale O accumulators
        for (int e = 0; e < 8; e++) {
            float a = sa[e + half * 8];
            for (int nt = 0; nt < 8; nt++) oacc[nt][e] *= a;
        }
        // P @ V  (V = kv part of k_eff, cols [wave*128, wave*128+128))
        for (int s = 0; s < 2; s++) {
            bf16x16 pf;
            for (int j = 0; j < 16; j++) {
                int kkl = half * 16 + j;
                pf[j] = sP[lp][s * 32 + kkl];
            }
            for (int nt = 0; nt < 8; nt++) {
                bf16x16 vf;
                const bf16* vbase = kEff + ((size_t)b * S_ + ktb + s * 32 + half * 16) * KEFF_
                                  + wave * 128 + nt * 16 + lp;
                for (int j = 0; j < 16; j++) vf[j] = vbase[(size_t)j * KEFF_];
                oacc[nt] = __builtin_amdgcn_wmma_f32_16x16x32_bf16(
                    false, pf, false, vf, (short)0, oacc[nt], false, false);
            }
        }
    }
    __syncthreads();
    for (int nt = 0; nt < 8; nt++)
        for (int e = 0; e < 8; e++) {
            int r = e + half * 8;
            float linv = 1.0f / sl[r];
            oLat[((size_t)(b * NH_ + h) * S_ + qt * 16 + r) * KVR_ + wave * 128 + nt * 16 + lp]
                = (bf16)(oacc[nt][e] * linv);
        }
}

// ---------------------------------------------------------------------------
// host launcher
// ---------------------------------------------------------------------------
extern "C" void kernel_launch(void* const* d_in, const int* in_sizes, int n_in,
                              void* d_out, int out_size, void* d_ws, size_t ws_size,
                              hipStream_t stream) {
    const float* x      = (const float*)d_in[0];
    const float* wq     = (const float*)d_in[1];
    const float* wkv_a  = (const float*)d_in[2];
    const float* kvnw   = (const float*)d_in[3];
    const float* wkv_b  = (const float*)d_in[4];
    const float* wo     = (const float*)d_in[5];
    const float* fcos   = (const float*)d_in[6];
    const float* fsin   = (const float*)d_in[7];
    float* out = (float*)d_out;

    const long ROWS = (long)B_ * S_;               // 4096
    char* ws = (char*)d_ws;
    size_t off = 0;
    auto alloc = [&](size_t bytes) -> void* {
        off = (off + 255) & ~(size_t)255;
        void* p = ws + off;
        off += bytes;
        return p;
    };

    bf16*  xBf    = (bf16*)alloc(ROWS * DIM_ * 2);
    bf16*  wqBf   = (bf16*)alloc((size_t)NH_ * QKHD_ * DIM_ * 2);
    bf16*  wkvaBf = (bf16*)alloc((size_t)KEFF_ * DIM_ * 2);
    bf16*  wkvbBf = (bf16*)alloc((size_t)NH_ * (NOPE_ + VHD_) * KVR_ * 2);
    bf16*  woBf   = (bf16*)alloc((size_t)DIM_ * NH_ * VHD_ * 2);
    float* qF     = (float*)alloc(ROWS * (size_t)(NH_ * QKHD_) * 4);
    bf16*  qBf    = (bf16*)alloc(ROWS * (size_t)(NH_ * QKHD_) * 2);
    float* kvF    = (float*)alloc(ROWS * (size_t)KEFF_ * 4);
    bf16*  kEff   = (bf16*)alloc(ROWS * (size_t)KEFF_ * 2);
    bf16*  qEff   = (bf16*)alloc((size_t)B_ * NH_ * S_ * KEFF_ * 2);
    bf16*  oLat   = (bf16*)alloc((size_t)B_ * NH_ * S_ * KVR_ * 2);
    bf16*  uBf    = (bf16*)alloc(ROWS * (size_t)DIM_ * 2);

    auto castN = [&](const float* src, bf16* dst, long n) {
        int blocks = (int)((n + 255) / 256);
        if (blocks > 8192) blocks = 8192;
        cast_f32_bf16<<<blocks, 256, 0, stream>>>(src, dst, n);
    };
    castN(x,     xBf,    ROWS * (long)DIM_);
    castN(wq,    wqBf,   (long)NH_ * QKHD_ * DIM_);
    castN(wkv_a, wkvaBf, (long)KEFF_ * DIM_);
    castN(wkv_b, wkvbBf, (long)NH_ * (NOPE_ + VHD_) * KVR_);
    castN(wo,    woBf,   (long)DIM_ * NH_ * VHD_);

    dim3 blk(128);

    // kv = x @ wkv_a^T  -> kvF (4096 x 576)
    wmma_gemm<<<dim3(KEFF_ / 64, ROWS / 64, 1), blk, 0, stream>>>(
        xBf, wkvaBf, kvF, (int)ROWS, KEFF_, DIM_,
        DIM_, DIM_, KEFF_, 0, 0, 0, 0, 0, 0, 1, 1, 0);

    kv_finish<<<dim3((int)ROWS), blk, 0, stream>>>(kvF, kvnw, fcos, fsin, kEff);

    // q = x @ wq^T -> qF (4096 x 3072)
    wmma_gemm<<<dim3(NH_ * QKHD_ / 64, ROWS / 64, 1), blk, 0, stream>>>(
        xBf, wqBf, qF, (int)ROWS, NH_ * QKHD_, DIM_,
        DIM_, DIM_, NH_ * QKHD_, 0, 0, 0, 0, 0, 0, 1, 1, 0);

    q_finish<<<dim3((int)ROWS, NH_), dim3(64), 0, stream>>>(qF, fcos, fsin, qBf, qEff);

    // q_abs[b,h] = q_nope[b,:,h,:] @ wkv_b_nope[h]  (NN)  -> qEff[...,0:512]
    wmma_gemm<<<dim3(KVR_ / 64, S_ / 64, B_ * NH_), blk, 0, stream>>>(
        qBf, wkvbBf, qEff, S_, KVR_, NOPE_,
        NH_ * QKHD_, KVR_, KEFF_,
        (long)S_ * NH_ * QKHD_, (long)QKHD_,
        0, (long)(NOPE_ + VHD_) * KVR_,
        (long)NH_ * S_ * KEFF_, (long)S_ * KEFF_,
        NH_, 0, 1);

    // flash attention over latent cache
    mla_attn<<<dim3(S_ / 16, NH_, B_), blk, 0, stream>>>(qEff, kEff, oLat);

    // u[b,:,h*128:(h+1)*128] = oLat[b,h] @ wkv_b_v[h]^T  (NT)
    wmma_gemm<<<dim3(VHD_ / 64, S_ / 64, B_ * NH_), blk, 0, stream>>>(
        oLat, wkvbBf + (size_t)NOPE_ * KVR_, uBf, S_, VHD_, KVR_,
        KVR_, KVR_, DIM_,
        (long)NH_ * S_ * KVR_, (long)S_ * KVR_,
        0, (long)(NOPE_ + VHD_) * KVR_,
        (long)S_ * DIM_, (long)VHD_,
        NH_, 1, 1);

    // out = u @ wo^T  (fp32 output)
    wmma_gemm<<<dim3(DIM_ / 64, ROWS / 64, 1), blk, 0, stream>>>(
        uBf, woBf, out, (int)ROWS, DIM_, DIM_,
        DIM_, DIM_, DIM_, 0, 0, 0, 0, 0, 0, 1, 1, 0);
}
